// TokensChooseScatterRouter_22428319220048
// MI455X (gfx1250) — matrix operand order repaired and checked by reading, not compiled, not run on hardware
//
#include <hip/hip_runtime.h>
#include <hip/hip_bf16.h>
#include <stdint.h>

// ---------------------------------------------------------------------------
// MoE TokensChooseScatterRouter for MI455X (gfx1250, wave32, WMMA).
// G=8 groups, T=4096 tokens/group, H=2048 hidden, E=64 experts, k=2.
// ---------------------------------------------------------------------------

#define G_ 8
#define T_ 4096
#define H_ 2048
#define E_ 64
#define GT_ (G_ * T_)

typedef __attribute__((ext_vector_type(16))) __bf16          v16bf;
typedef __attribute__((ext_vector_type(8)))  float           v8f;
typedef __attribute__((ext_vector_type(4)))  float           v4f;
typedef __attribute__((ext_vector_type(16))) unsigned short  v16us;
typedef __attribute__((ext_vector_type(8)))  unsigned short  v8us;

__device__ __forceinline__ unsigned short f2bf(float f) {
    // round-to-nearest-even f32 -> bf16 (bit-exact, used on the tiny w matrix)
    unsigned u = __builtin_bit_cast(unsigned, f);
    u += 0x7FFFu + ((u >> 16) & 1u);
    return (unsigned short)(u >> 16);
}

// ---------------------------------------------------------------------------
// Kernel P: w [H,E] f32  ->  w_t [E,H] bf16 (so B fragments are contiguous)
// ---------------------------------------------------------------------------
__global__ __launch_bounds__(256) void wt_bf16_kernel(
    const float* __restrict__ w, unsigned short* __restrict__ wt) {
    int idx = blockIdx.x * 256 + threadIdx.x;  // H*E = 131072 elements
    int e = idx & (E_ - 1);
    int h = idx >> 6;
    wt[(size_t)e * H_ + h] = f2bf(w[(size_t)h * E_ + e]);
}

// ---------------------------------------------------------------------------
// Kernel A: logits[GT,64] = x[GT,2048] @ w_t^T + bias  via v_wmma_f32_16x16x32_bf16
// One wave = 16-token M-tile x all 64 experts (4 N-tiles, 4 accumulators).
// Software-pipelined (2-deep ping-pong) so A/B loads stay in flight across the
// WMMAs instead of s_wait_loadcnt 0x0 stalls. A-stream loads are non-temporal
// (read-once 256MB stream); w_t stays L2/WGP$-resident.
//
// A-frag per lane (16x32 bf16): lanes 0-15 row M=lane, K = k0+[0..7], k0+16+[0..7];
// lanes 16-31 same row, K = k0+8+[0..7], k0+24+[0..7].
// B-frag per lane (32x16 bf16): col N = lane%16, K = k0 + (lane/16)*8 + [0..7], k0 + 16 + (lane/16)*8 + [0..7].
// ---------------------------------------------------------------------------
__global__ __launch_bounds__(256) void router_gemm_wmma(
    const float* __restrict__ x, const unsigned short* __restrict__ wt,
    const float* __restrict__ bias, float* __restrict__ logits) {
    const int tid  = threadIdx.x;
    const int lane = tid & 31;
    const int wv   = tid >> 5;
    const int half = lane >> 4;
    const int lrow = lane & 15;
    const int m0   = (blockIdx.x * 8 + wv) * 16;

    v8f acc[4] = {};
    const float* arow = x + (size_t)(m0 + lrow) * H_;

    auto loadA = [&](int k0, v4f (&dst)[4]) {
        const float* ap = arow + k0 + half * 8;
        dst[0] = __builtin_nontemporal_load((const v4f*)(ap));
        dst[1] = __builtin_nontemporal_load((const v4f*)(ap + 4));
        dst[2] = __builtin_nontemporal_load((const v4f*)(ap + 16));
        dst[3] = __builtin_nontemporal_load((const v4f*)(ap + 20));
    };
    auto loadB = [&](int k0, v8us (&dst)[8]) {
        #pragma unroll
        for (int n = 0; n < 4; n++) {
            const unsigned short* bp =
                wt + (size_t)(n * 16 + lrow) * H_ + k0 + half * 8;
            dst[2 * n]     = *(const v8us*)(bp);
            dst[2 * n + 1] = *(const v8us*)(bp + 16);
        }
    };
    auto compute = [&](v4f (&af)[4], v8us (&bf)[8]) {
        v16bf a;  // native converts -> packed f32->bf16 cvt
        #pragma unroll
        for (int q = 0; q < 4; q++) {
            a[4 * q + 0] = (__bf16)af[q][0];
            a[4 * q + 1] = (__bf16)af[q][1];
            a[4 * q + 2] = (__bf16)af[q][2];
            a[4 * q + 3] = (__bf16)af[q][3];
        }
        #pragma unroll
        for (int n = 0; n < 4; n++) {
            v16us bu;
            #pragma unroll
            for (int i = 0; i < 8; i++) {
                bu[i] = bf[2 * n][i]; bu[8 + i] = bf[2 * n + 1][i];
            }
            acc[n] = __builtin_amdgcn_wmma_f32_16x16x32_bf16(
                false, a, false, __builtin_bit_cast(v16bf, bu),
                (short)0, acc[n], false, false);
        }
    };

    v4f  abuf0[4], abuf1[4];
    v8us bbuf0[8], bbuf1[8];
    loadA(0, abuf0);
    loadB(0, bbuf0);
    for (int k0 = 0; k0 < H_; k0 += 64) {
        loadA(k0 + 32, abuf1);          // prefetch stage 1
        loadB(k0 + 32, bbuf1);
        compute(abuf0, bbuf0);          // consume stage 0 (loads issued last iter)
        int kn = (k0 + 64 < H_) ? (k0 + 64) : 0;  // clamped dummy on last iter
        loadA(kn, abuf0);               // prefetch stage 0 of next iter
        loadB(kn, bbuf0);
        compute(abuf1, bbuf1);          // consume stage 1
    }

    // C/D layout: VGPR j holds M = m0 + j + half*8, N = n*16 + lrow.
    #pragma unroll
    for (int n = 0; n < 4; n++) {
        float bb = bias[n * 16 + lrow];
        #pragma unroll
        for (int j = 0; j < 8; j++) {
            int row = m0 + j + half * 8;
            logits[(size_t)row * E_ + n * 16 + lrow] = acc[n][j] + bb;
        }
    }
}

// ---------------------------------------------------------------------------
// Kernel B1: per-token softmax (writes router_probs), top-2 gates/indices,
// deterministic per-block partials for aux (sum of probs per expert) and
// z-loss (sum of logsumexp^2). One thread per token; 256 tokens per block,
// each block entirely within one group (4096 % 256 == 0).
// ---------------------------------------------------------------------------
__global__ __launch_bounds__(256) void softmax_top2(
    const float* __restrict__ logits, float* __restrict__ probs_out,
    float* __restrict__ gate0, float* __restrict__ gate1,
    int* __restrict__ idx0, int* __restrict__ idx1,
    float* __restrict__ spart, float* __restrict__ zpart) {
    __shared__ float ls[8 * 64];
    __shared__ float zs[8];
    const int tid = threadIdx.x, lane = tid & 31, wv = tid >> 5;
    const int t = blockIdx.x * 256 + tid;  // global token id
    const float* lp = logits + (size_t)t * E_;

    float l[64];
    #pragma unroll
    for (int i = 0; i < 16; i++) {
        v4f v = *(const v4f*)(lp + 4 * i);
        l[4 * i] = v[0]; l[4 * i + 1] = v[1]; l[4 * i + 2] = v[2]; l[4 * i + 3] = v[3];
    }
    float mx = l[0];
    #pragma unroll
    for (int e = 1; e < 64; e++) mx = fmaxf(mx, l[e]);
    float s = 0.f;
    #pragma unroll
    for (int e = 0; e < 64; e++) { float pe = __expf(l[e] - mx); s += pe; l[e] = pe; }
    float logz = mx + __logf(s);
    float inv  = 1.f / s;

    float g0 = -1.f, g1 = -1.f; int i0 = 0, i1 = 0;
    float* po = probs_out + (size_t)t * E_;
    #pragma unroll
    for (int e = 0; e < 64; e++) {
        float p = l[e] * inv;
        po[e] = p;
        if (p > g0)      { g1 = g0; i1 = i0; g0 = p; i0 = e; }
        else if (p > g1) { g1 = p; i1 = e; }
        float ps = p;  // cross-lane (32-token) sum for this expert
        #pragma unroll
        for (int m = 16; m >= 1; m >>= 1) ps += __shfl_xor(ps, m, 32);
        if (lane == 0) ls[wv * 64 + e] = ps;
    }
    float z2 = logz * logz;
    #pragma unroll
    for (int m = 16; m >= 1; m >>= 1) z2 += __shfl_xor(z2, m, 32);
    if (lane == 0) zs[wv] = z2;

    gate0[t] = g0; gate1[t] = g1; idx0[t] = i0; idx1[t] = i1;
    __syncthreads();
    if (tid < 64) {
        float a = 0.f;
        for (int w = 0; w < 8; w++) a += ls[w * 64 + tid];
        spart[blockIdx.x * 64 + tid] = a;
    }
    if (tid == 0) {
        float a = 0.f;
        for (int w = 0; w < 8; w++) a += zs[w];
        zpart[blockIdx.x] = a;
    }
}

// ---------------------------------------------------------------------------
// Kernel B2: per group — batch-prioritized ordering (bitonic sort on
// (inv_gate_bits, token_id) == stable argsort(-gate0)), k-major cumulative
// per-expert position counting over 8192 slots, capacity drop, outputs.
// One block per group, 1024 threads (32 waves), 40KB static LDS.
// ---------------------------------------------------------------------------
__global__ __launch_bounds__(1024) void route_priority(
    const float* __restrict__ gate0, const float* __restrict__ gate1,
    const int* __restrict__ idx0, const int* __restrict__ idx1,
    int* __restrict__ cnt_out, const int* __restrict__ cap_ptr,
    float* __restrict__ out_di, float* __restrict__ out_cw) {
    __shared__ __align__(16) unsigned char smem[40960];
    unsigned long long* keys    = (unsigned long long*)smem;            // [4096] (phase 1-2)
    unsigned short*     ordering= (unsigned short*)smem;                // [4096] (phase 3+)
    unsigned short*     rank_of = (unsigned short*)(smem + 8192);       // [4096]
    unsigned short*     lrank   = (unsigned short*)(smem + 16384);      // [8192]
    int*                hist    = (int*)(smem + 32768);                 // [32][64]

    const int g = blockIdx.x, tid = threadIdx.x, lane = tid & 31, wv = tid >> 5;
    const int cap  = *cap_ptr;
    const int base = g * T_;

    // Phase 1: composite sort keys. Ascending order == descending gate, ties by id.
    for (int t = tid; t < T_; t += 1024) {
        unsigned gb = __builtin_bit_cast(unsigned, gate0[base + t]);  // gate > 0
        keys[t] = ((unsigned long long)(0xFFFFFFFFu - gb) << 32) | (unsigned)t;
    }
    __syncthreads();

    // Phase 2: bitonic sort, 4096 keys, ascending.
    for (unsigned size = 2; size <= 4096; size <<= 1) {
        for (unsigned stride = size >> 1; stride > 0; stride >>= 1) {
            for (unsigned i = tid; i < 2048; i += 1024) {
                unsigned pos = 2 * i - (i & (stride - 1));
                unsigned oth = pos + stride;
                bool asc = ((pos & size) == 0);
                unsigned long long a = keys[pos], b = keys[oth];
                if ((a > b) == asc) { keys[pos] = b; keys[oth] = a; }
            }
            __syncthreads();
        }
    }

    // Phase 3: ordering / inverse ordering (overlays key storage).
    unsigned myid[4], myr[4];
    {
        int c = 0;
        for (int r = tid; r < T_; r += 1024) {
            myid[c] = (unsigned)(keys[r] & 0xFFFFFFFFu);
            myr[c] = (unsigned)r; c++;
        }
    }
    __syncthreads();
    #pragma unroll
    for (int c = 0; c < 4; c++) {
        ordering[myr[c]] = (unsigned short)myid[c];
        rank_of[myid[c]] = (unsigned short)myr[c];
    }
    for (int i = tid; i < 32 * 64; i += 1024) hist[i] = 0;
    __syncthreads();

    // Phase 4: wave-chunked local ranks over the k-major slot sequence.
    // Slot s in [0,8192): rank r = s & 4095, k = s >> 12 (chunks never straddle k).
    for (int rr = 0; rr < 8; rr++) {
        int s  = wv * 256 + rr * 32 + lane;
        int r  = s & (T_ - 1);
        int kk = s >> 12;
        int tok = ordering[r];
        int e = kk ? idx1[base + tok] : idx0[base + tok];
        int before = 0, total = 0;
        for (int j = 0; j < 32; j++) {
            int ej = __shfl(e, j, 32);
            if (ej == e) { total++; if (j < lane) before++; }
        }
        int hb = hist[wv * 64 + e];
        lrank[s] = (unsigned short)(hb + before);
        if (before == total - 1) hist[wv * 64 + e] = hb + total;  // one writer per expert
    }
    __syncthreads();

    // Phase 5: exclusive scan of per-wave histograms -> chunk base offsets.
    // Final running totals are exactly the per-(g,e) expert_mask counts (top-2
    // indices are always distinct), reused by the aux loss.
    if (tid < 64) {
        int run = 0;
        for (int w = 0; w < 32; w++) {
            int v = hist[w * 64 + tid];
            hist[w * 64 + tid] = run;
            run += v;
        }
        cnt_out[g * 64 + tid] = run;
    }
    __syncthreads();

    // Phase 6: gather back to original token order, capacity drop, write outputs.
    for (int t = tid; t < T_; t += 1024) {
        int gt = base + t;
        int r  = rank_of[t];
        int e0 = idx0[gt], e1 = idx1[gt];
        int s0 = r, s1 = T_ + r;
        int p0 = hist[(s0 >> 8) * 64 + e0] + (int)lrank[s0];
        int p1 = hist[(s1 >> 8) * 64 + e1] + (int)lrank[s1];
        float w0 = gate0[gt] * (p0 < cap ? 1.f : 0.f);
        float w1 = gate1[gt] * (p1 < cap ? 1.f : 0.f);
        float* di = out_di + (size_t)gt * 4;
        di[0] = (float)e0; di[1] = (float)p0;
        di[2] = (float)e1; di[3] = (float)p1;
        out_cw[(size_t)gt * 2 + 0] = w0;
        out_cw[(size_t)gt * 2 + 1] = w1;
    }
}

// ---------------------------------------------------------------------------
// Kernel C: deterministic fixed-order reduction of the two scalar losses.
// aux = E^2 * mean_{g,e}[(cnt/T)*(sumprob/T)] = (E/(G*T*T)) * sum(cnt*sumprob)
// ---------------------------------------------------------------------------
__global__ void finalize_losses(
    const float* __restrict__ spart, const float* __restrict__ zpart,
    const int* __restrict__ cnt, float* __restrict__ out_aux,
    float* __restrict__ out_z) {
    if (threadIdx.x == 0 && blockIdx.x == 0) {
        float z = 0.f;
        for (int b = 0; b < 128; b++) z += zpart[b];
        *out_z = z / (float)(G_ * T_);
        double acc = 0.0;
        for (int g = 0; g < G_; g++)
            for (int e = 0; e < E_; e++) {
                float sp = 0.f;
                for (int b = 0; b < 16; b++) sp += spart[(g * 16 + b) * 64 + e];
                acc += (double)cnt[g * 64 + e] * (double)sp;
            }
        *out_aux = (float)(acc * (double)E_ /
                           ((double)G_ * (double)T_ * (double)T_));
    }
}

// ---------------------------------------------------------------------------
// Workspace layout (bytes)
// ---------------------------------------------------------------------------
#define WS_WT     0         // 64*2048*2      = 262144   bf16 w^T
#define WS_LOGITS 262144    // 32768*64*4     = 8388608
#define WS_GATE0  8650752   // 32768*4
#define WS_GATE1  8781824
#define WS_IDX0   8912896
#define WS_IDX1   9043968
#define WS_SPART  9175040   // 128*64*4
#define WS_ZPART  9207808   // 128*4
#define WS_CNT    9208320   // 8*64*4

extern "C" void kernel_launch(void* const* d_in, const int* in_sizes, int n_in,
                              void* d_out, int out_size, void* d_ws, size_t ws_size,
                              hipStream_t stream) {
    const float* x    = (const float*)d_in[0];  // [8,4096,2048]
    const float* w    = (const float*)d_in[1];  // [2048,64]
    const float* bias = (const float*)d_in[2];  // [64]
    const int*   cap  = (const int*)d_in[3];    // scalar 160

    uint8_t* ws = (uint8_t*)d_ws;
    unsigned short* wt = (unsigned short*)(ws + WS_WT);
    float* logits = (float*)(ws + WS_LOGITS);
    float* gate0  = (float*)(ws + WS_GATE0);
    float* gate1  = (float*)(ws + WS_GATE1);
    int*   idx0   = (int*)(ws + WS_IDX0);
    int*   idx1   = (int*)(ws + WS_IDX1);
    float* spart  = (float*)(ws + WS_SPART);
    float* zpart  = (float*)(ws + WS_ZPART);
    int*   cnt    = (int*)(ws + WS_CNT);

    // Output tuple, flat: dispatch[8,4096,2,2], combine[8,4096,2], aux,
    // probs[8,4096,64], z_loss
    float* out       = (float*)d_out;
    float* out_di    = out;                     // 131072
    float* out_cw    = out + 131072;            // 65536
    float* out_aux   = out + 196608;            // 1
    float* out_probs = out + 196609;            // 2097152
    float* out_z     = out + 2293761;           // 1

    wt_bf16_kernel<<<512, 256, 0, stream>>>(w, wt);
    router_gemm_wmma<<<256, 256, 0, stream>>>(x, wt, bias, logits);
    softmax_top2<<<128, 256, 0, stream>>>(logits, out_probs, gate0, gate1,
                                          idx0, idx1, spart, zpart);
    route_priority<<<8, 1024, 0, stream>>>(gate0, gate1, idx0, idx1, cnt, cap,
                                           out_di, out_cw);
    finalize_losses<<<1, 64, 0, stream>>>(spart, zpart, cnt, out_aux, out_z);
}
